// TransformerDecoderLayer_38603166056860
// MI455X (gfx1250) — matrix operand order repaired and checked by reading, not compiled, not run on hardware
//
#include <hip/hip_runtime.h>

// ===================== types & helpers =====================
typedef unsigned short u16;
typedef unsigned int   u32;
typedef __attribute__((ext_vector_type(16))) __bf16 v16bf;
typedef __attribute__((ext_vector_type(8)))  float  v8f;

union FragBF { v16bf v; u32 u[8]; u16 s[16]; };
union AccF   { v8f v; float f[8]; };

__device__ __forceinline__ u16 f2bf(float x) {
  u32 u = __builtin_bit_cast(u32, x);
  return (u16)((u + 0x7FFFu + ((u >> 16) & 1u)) >> 16);   // round-to-nearest-even
}
__device__ __forceinline__ float bf2f(u16 h) {
  u32 u = ((u32)h) << 16;
  return __builtin_bit_cast(float, u);
}
__device__ __forceinline__ v8f wmma_bf16(v16bf a, v16bf b, v8f c) {
  return __builtin_amdgcn_wmma_f32_16x16x32_bf16(false, a, false, b, (short)0, c, false, false);
}
// K-offset of the bf16 pair held in dword v of a 16-bit A fragment (ISA 7.12.2)
__device__ __forceinline__ int abase(int v) { return ((v & 3) << 1) + ((v >> 2) << 4); }

// Async copy 16B/lane from global (saddr + 32-bit voffset) straight into LDS (ASYNCcnt).
__device__ __forceinline__ void async_g2lds_b128(const void* sbase, u32 voff_bytes, u32 lds_byte_addr) {
  asm volatile("global_load_async_to_lds_b128 %0, %1, %2 offset:0"
               :: "v"(lds_byte_addr), "v"(voff_bytes), "s"(sbase)
               : "memory");
}
__device__ __forceinline__ void wait_async0() {
  asm volatile("s_wait_asynccnt 0" ::: "memory");
}
// Low 32 bits of a generic pointer to LDS == LDS byte offset (aperture addressing).
__device__ __forceinline__ u32 lds_addr(const void* p) {
  return (u32)(size_t)p;
}

// ===================== fused group RMSNorm (f32 -> bf16) =====================
// H=2048, G=256 -> 8 groups/token; one 256-thread block per token, wave w = group w.
__global__ __launch_bounds__(256) void rmsnorm_kernel(const float* __restrict__ x,
                                                      const float* __restrict__ w,
                                                      u16* __restrict__ out) {
  int lane = threadIdx.x & 31, wv = threadIdx.x >> 5;
  size_t base = (size_t)blockIdx.x * 2048 + wv * 256 + lane * 8;
  float4 a = *(const float4*)(x + base);
  float4 b = *(const float4*)(x + base + 4);
  float ss = a.x*a.x + a.y*a.y + a.z*a.z + a.w*a.w
           + b.x*b.x + b.y*b.y + b.z*b.z + b.w*b.w;
#pragma unroll
  for (int m = 1; m < 32; m <<= 1) ss += __shfl_xor(ss, m, 32);
  float rinv = 1.0f / sqrtf(ss * (1.0f / 256.0f) + 1e-5f);
  const float* wp = w + wv * 256 + lane * 8;
  float xv[8] = {a.x, a.y, a.z, a.w, b.x, b.y, b.z, b.w};
  u16 r[8];
#pragma unroll
  for (int i = 0; i < 8; i++) r[i] = f2bf(xv[i] * rinv * wp[i]);
  uint4 o;
  o.x = r[0] | ((u32)r[1] << 16); o.y = r[2] | ((u32)r[3] << 16);
  o.z = r[4] | ((u32)r[5] << 16); o.w = r[6] | ((u32)r[7] << 16);
  *(uint4*)(out + base) = o;
}

// ===================== generic WMMA GEMM: C[M,N] = A[M,K] * B[N,K]^T =====================
// A: bf16 (workspace) -> staged with GLOBAL_LOAD_ASYNC_TO_LDS_B128 (double-buffered).
// B: fp32 weights -> register-prefetched, converted to bf16, stored to LDS.
// EPI: 0 = bf16 store, 1 = bf16 transposed store [b,n,s] (V), 2 = f32 resid+acc, 3 = f32 +=
constexpr int BM = 128, BN = 128, BK = 32, LDT = 40;   // LDT: padded LDS row (bf16 elems)

template <int EPI>
__global__ __launch_bounds__(256) void gemm_kernel(const u16* __restrict__ A,
                                                   const float* __restrict__ B,
                                                   void* __restrict__ Out,
                                                   const float* __restrict__ Resid,
                                                   int M, int N, int K, int Sdim) {
  __shared__ alignas(16) u16 As[2][BM * LDT];
  __shared__ alignas(16) u16 Bs[2][BN * LDT];
  int tid = threadIdx.x;
  int lane = tid & 31, wv = tid >> 5;
  int half = lane >> 4, l16 = lane & 15;
  int wm0 = (wv & 3) * 32, wn0 = (wv >> 2) * 64;    // 4x2 wave grid: 32x64 per wave
  int bm = blockIdx.y * BM, bn = blockIdx.x * BN;
  AccF acc[2][4];
#pragma unroll
  for (int mi = 0; mi < 2; mi++)
#pragma unroll
    for (int ni = 0; ni < 4; ni++)
#pragma unroll
      for (int r = 0; r < 8; r++) acc[mi][ni].f[r] = 0.0f;

  float4 breg[4];
  auto stage_a = [&](int k0, int buf) {              // 512 x 16B chunks, async -> LDS
#pragma unroll
    for (int i = 0; i < 2; i++) {
      int c = tid + i * 256;
      int row = c >> 2, k8 = (c & 3) << 3;
      u32 goff = (u32)(((size_t)(bm + row) * K + k0 + k8) * 2);
      async_g2lds_b128(A, goff, lds_addr(&As[buf][row * LDT + k8]));
    }
  };
  auto load_b = [&](int k0) {                        // 1024 x float4 chunks -> regs
#pragma unroll
    for (int i = 0; i < 4; i++) {
      int c = tid + i * 256;
      int row = c >> 3, k4 = (c & 7) << 2;
      breg[i] = *(const float4*)(B + (size_t)(bn + row) * K + k0 + k4);
    }
  };
  auto store_b = [&](int buf) {                      // cvt f32 -> bf16, regs -> LDS
#pragma unroll
    for (int i = 0; i < 4; i++) {
      int c = tid + i * 256;
      int row = c >> 3, k4 = (c & 7) << 2;
      uint2 p;
      p.x = f2bf(breg[i].x) | ((u32)f2bf(breg[i].y) << 16);
      p.y = f2bf(breg[i].z) | ((u32)f2bf(breg[i].w) << 16);
      *(uint2*)(&Bs[buf][row * LDT + k4]) = p;
    }
  };

  stage_a(0, 0);
  load_b(0);
  store_b(0);
  wait_async0();
  __syncthreads();

  int buf = 0;
  for (int k0 = 0; k0 < K; k0 += BK) {
    int nxt = buf ^ 1;
    bool hasnext = (k0 + BK) < K;
    if (hasnext) {                                   // overlap next-tile fill with WMMAs
      stage_a(k0 + BK, nxt);
      load_b(k0 + BK);
    }
    FragBF af[2], bfr[4];
#pragma unroll
    for (int mi = 0; mi < 2; mi++) {
      int mr = wm0 + mi * 16 + l16;
#pragma unroll
      for (int v = 0; v < 8; v++)
        af[mi].u[v] = *(const u32*)(&As[buf][mr * LDT + abase(v) + half * 8]);
    }
#pragma unroll
    for (int ni = 0; ni < 4; ni++) {
      int nr = wn0 + ni * 16 + l16;
#pragma unroll
      for (int v = 0; v < 8; v++)
        bfr[ni].u[v] = *(const u32*)(&Bs[buf][nr * LDT + abase(v) + half * 8]);
    }
#pragma unroll
    for (int mi = 0; mi < 2; mi++)
#pragma unroll
      for (int ni = 0; ni < 4; ni++)
        acc[mi][ni].v = wmma_bf16(af[mi].v, bfr[ni].v, acc[mi][ni].v);
    if (hasnext) {
      store_b(nxt);
      wait_async0();
      __syncthreads();
    }
    buf = nxt;
  }
#pragma unroll
  for (int mi = 0; mi < 2; mi++)
#pragma unroll
    for (int ni = 0; ni < 4; ni++)
#pragma unroll
      for (int r = 0; r < 8; r++) {
        int m = bm + wm0 + mi * 16 + r + half * 8;
        int n = bn + wn0 + ni * 16 + l16;
        float val = acc[mi][ni].f[r];
        if (EPI == 0) {
          ((u16*)Out)[(size_t)m * N + n] = f2bf(val);
        } else if (EPI == 1) {
          int b = m / Sdim, s = m - b * Sdim;
          ((u16*)Out)[((size_t)b * N + n) * Sdim + s] = f2bf(val);
        } else if (EPI == 2) {
          ((float*)Out)[(size_t)m * N + n] = Resid[(size_t)m * N + n] + val;
        } else {
          ((float*)Out)[(size_t)m * N + n] += val;
        }
      }
}

// ===================== fused gate/up GEMM with SiLU epilogue =====================
__global__ __launch_bounds__(256) void gemm_gateup_kernel(const u16* __restrict__ A,
                                                          const float* __restrict__ Bg,
                                                          const float* __restrict__ Bu,
                                                          u16* __restrict__ Act,
                                                          int M, int N, int K) {
  __shared__ alignas(16) u16 As[2][BM * LDT];
  __shared__ alignas(16) u16 Bgs[2][BN * LDT];
  __shared__ alignas(16) u16 Bus[2][BN * LDT];
  int tid = threadIdx.x;
  int lane = tid & 31, wv = tid >> 5;
  int half = lane >> 4, l16 = lane & 15;
  int wm0 = (wv & 3) * 32, wn0 = (wv >> 2) * 64;
  int bm = blockIdx.y * BM, bn = blockIdx.x * BN;
  AccF ag[2][4], au[2][4];
#pragma unroll
  for (int mi = 0; mi < 2; mi++)
#pragma unroll
    for (int ni = 0; ni < 4; ni++)
#pragma unroll
      for (int r = 0; r < 8; r++) { ag[mi][ni].f[r] = 0.0f; au[mi][ni].f[r] = 0.0f; }

  float4 bgreg[4], bureg[4];
  auto stage_a = [&](int k0, int buf) {
#pragma unroll
    for (int i = 0; i < 2; i++) {
      int c = tid + i * 256;
      int row = c >> 2, k8 = (c & 3) << 3;
      u32 goff = (u32)(((size_t)(bm + row) * K + k0 + k8) * 2);
      async_g2lds_b128(A, goff, lds_addr(&As[buf][row * LDT + k8]));
    }
  };
  auto load_b = [&](int k0) {
#pragma unroll
    for (int i = 0; i < 4; i++) {
      int c = tid + i * 256;
      int row = c >> 3, k4 = (c & 7) << 2;
      bgreg[i] = *(const float4*)(Bg + (size_t)(bn + row) * K + k0 + k4);
      bureg[i] = *(const float4*)(Bu + (size_t)(bn + row) * K + k0 + k4);
    }
  };
  auto store_b = [&](int buf) {
#pragma unroll
    for (int i = 0; i < 4; i++) {
      int c = tid + i * 256;
      int row = c >> 3, k4 = (c & 7) << 2;
      uint2 pg, pu;
      pg.x = f2bf(bgreg[i].x) | ((u32)f2bf(bgreg[i].y) << 16);
      pg.y = f2bf(bgreg[i].z) | ((u32)f2bf(bgreg[i].w) << 16);
      pu.x = f2bf(bureg[i].x) | ((u32)f2bf(bureg[i].y) << 16);
      pu.y = f2bf(bureg[i].z) | ((u32)f2bf(bureg[i].w) << 16);
      *(uint2*)(&Bgs[buf][row * LDT + k4]) = pg;
      *(uint2*)(&Bus[buf][row * LDT + k4]) = pu;
    }
  };

  stage_a(0, 0);
  load_b(0);
  store_b(0);
  wait_async0();
  __syncthreads();

  int buf = 0;
  for (int k0 = 0; k0 < K; k0 += BK) {
    int nxt = buf ^ 1;
    bool hasnext = (k0 + BK) < K;
    if (hasnext) {
      stage_a(k0 + BK, nxt);
      load_b(k0 + BK);
    }
    FragBF af[2], bfr[4];
#pragma unroll
    for (int mi = 0; mi < 2; mi++) {
      int mr = wm0 + mi * 16 + l16;
#pragma unroll
      for (int v = 0; v < 8; v++)
        af[mi].u[v] = *(const u32*)(&As[buf][mr * LDT + abase(v) + half * 8]);
    }
#pragma unroll
    for (int ni = 0; ni < 4; ni++) {
      int nr = wn0 + ni * 16 + l16;
#pragma unroll
      for (int v = 0; v < 8; v++)
        bfr[ni].u[v] = *(const u32*)(&Bgs[buf][nr * LDT + abase(v) + half * 8]);
    }
#pragma unroll
    for (int mi = 0; mi < 2; mi++)
#pragma unroll
      for (int ni = 0; ni < 4; ni++)
        ag[mi][ni].v = wmma_bf16(af[mi].v, bfr[ni].v, ag[mi][ni].v);
#pragma unroll
    for (int ni = 0; ni < 4; ni++) {
      int nr = wn0 + ni * 16 + l16;
#pragma unroll
      for (int v = 0; v < 8; v++)
        bfr[ni].u[v] = *(const u32*)(&Bus[buf][nr * LDT + abase(v) + half * 8]);
    }
#pragma unroll
    for (int mi = 0; mi < 2; mi++)
#pragma unroll
      for (int ni = 0; ni < 4; ni++)
        au[mi][ni].v = wmma_bf16(af[mi].v, bfr[ni].v, au[mi][ni].v);
    if (hasnext) {
      store_b(nxt);
      wait_async0();
      __syncthreads();
    }
    buf = nxt;
  }
#pragma unroll
  for (int mi = 0; mi < 2; mi++)
#pragma unroll
    for (int ni = 0; ni < 4; ni++)
#pragma unroll
      for (int r = 0; r < 8; r++) {
        int m = bm + wm0 + mi * 16 + r + half * 8;
        int n = bn + wn0 + ni * 16 + l16;
        float g = ag[mi][ni].f[r];
        float u = au[mi][ni].f[r];
        float s = g / (1.0f + __expf(-g));            // SiLU
        Act[(size_t)m * N + n] = f2bf(s * u);
      }
}

// ===================== RoPE (in-place on bf16 [tok][NH][128]) =====================
__global__ void rope_kernel(u16* __restrict__ buf, int NH, int total) {
  int idx = blockIdx.x * 256 + threadIdx.x;
  if (idx >= total) return;
  int d = idx & 63;
  int t = idx >> 6;
  int hh = t % NH;
  int tok = t / NH;
  int s = tok & 2047;                                  // S = 2048
  float inv = __expf(-(float)d * (9.210340371976184f / 64.0f));  // 10000^(-d/64)
  float ang = (float)s * inv;
  float cs, sn;
  sincosf(ang, &sn, &cs);
  size_t base = ((size_t)tok * NH + hh) * 128;
  float x1 = bf2f(buf[base + d]);
  float x2 = bf2f(buf[base + 64 + d]);
  buf[base + d]      = f2bf(x1 * cs - x2 * sn);
  buf[base + 64 + d] = f2bf(x2 * cs + x1 * sn);
}

// ===================== windowed flash attention (WMMA QK^T and PV) =====================
// q: bf16 [B,S,HQ,D]; k: bf16 [B,S,HKV,D]; vt: bf16 [B,HKV,D,S] (transposed); o: bf16 [B,S,HQ,D]
__global__ __launch_bounds__(128) void attn_kernel(const u16* __restrict__ q,
                                                   const u16* __restrict__ k,
                                                   const u16* __restrict__ vt,
                                                   u16* __restrict__ o) {
  constexpr int S = 2048, Dh = 128, HQn = 16, HKVn = 4, Wn = 1024;
  int lane = threadIdx.x & 31, wv = threadIdx.x >> 5;
  int half = lane >> 4, l16 = lane & 15;
  int i0 = blockIdx.x * 64 + wv * 16;                  // 16-query tile per wave
  int h = blockIdx.y, b = blockIdx.z;
  int hkv = h >> 2;                                    // rep = HQ/HKV = 4
  __shared__ alignas(16) u16 Pbuf[4][16 * 32];         // wave-private P transpose buffer
  u16* P = Pbuf[wv];

  FragBF qf[4];                                        // Q tile 16x128 = 4 K-chunks
  {
    size_t qrow = ((size_t)(b * S + i0 + l16) * HQn + h) * Dh;
#pragma unroll
    for (int c = 0; c < 4; c++)
#pragma unroll
      for (int v = 0; v < 8; v++)
        qf[c].u[v] = *(const u32*)(q + qrow + c * 32 + abase(v) + half * 8);
  }
  AccF ao[8];
#pragma unroll
  for (int c = 0; c < 8; c++)
#pragma unroll
    for (int r = 0; r < 8; r++) ao[c].f[r] = 0.0f;
  float rowmax[8], rowsum[8];
#pragma unroll
  for (int r = 0; r < 8; r++) { rowmax[r] = -3.0e38f; rowsum[r] = 0.0f; }

  int js = i0 - Wn + 16;
  if (js < 0) js = 0;
  js &= ~31;
  const float scale = 0.08838834764831845f;            // 1/sqrt(128)

  for (int j0 = js; j0 <= i0 + 15; j0 += 32) {
    AccF s0, s1;
#pragma unroll
    for (int r = 0; r < 8; r++) { s0.f[r] = 0.0f; s1.f[r] = 0.0f; }
    {
      size_t krow0 = ((size_t)(b * S + j0 + l16) * HKVn + hkv) * Dh;
      size_t krow1 = ((size_t)(b * S + j0 + 16 + l16) * HKVn + hkv) * Dh;
#pragma unroll
      for (int c = 0; c < 4; c++) {
        FragBF kf;
#pragma unroll
        for (int v = 0; v < 8; v++)
          kf.u[v] = *(const u32*)(k + krow0 + c * 32 + abase(v) + half * 8);
        s0.v = wmma_bf16(qf[c].v, kf.v, s0.v);
#pragma unroll
        for (int v = 0; v < 8; v++)
          kf.u[v] = *(const u32*)(k + krow1 + c * 32 + abase(v) + half * 8);
        s1.v = wmma_bf16(qf[c].v, kf.v, s1.v);
      }
    }
    float p0a[8], p1a[8], corr[8];
#pragma unroll
    for (int r = 0; r < 8; r++) {
      int i = i0 + r + half * 8;
      float v0 = s0.f[r] * scale;
      int jj = j0 + l16;
      if (jj > i || jj <= i - Wn) v0 = -3.0e38f;
      float v1 = s1.f[r] * scale;
      jj = j0 + 16 + l16;
      if (jj > i || jj <= i - Wn) v1 = -3.0e38f;
      float mx = fmaxf(v0, v1);
#pragma unroll
      for (int m = 1; m < 16; m <<= 1) mx = fmaxf(mx, __shfl_xor(mx, m, 32));
      float nm = fmaxf(rowmax[r], mx);
      float cr = __expf(rowmax[r] - nm);
      rowmax[r] = nm;
      float e0 = __expf(v0 - nm), e1 = __expf(v1 - nm);
      float ps = e0 + e1;
#pragma unroll
      for (int m = 1; m < 16; m <<= 1) ps += __shfl_xor(ps, m, 32);
      rowsum[r] = rowsum[r] * cr + ps;
      corr[r] = cr;
      p0a[r] = e0; p1a[r] = e1;
    }
#pragma unroll
    for (int c = 0; c < 8; c++)
#pragma unroll
      for (int r = 0; r < 8; r++) ao[c].f[r] *= corr[r];
    // C-layout -> A-layout transpose of P through wave-private LDS
#pragma unroll
    for (int r = 0; r < 8; r++) {
      int m = r + half * 8;
      P[m * 32 + l16]      = f2bf(p0a[r]);
      P[m * 32 + 16 + l16] = f2bf(p1a[r]);
    }
    asm volatile("s_wait_dscnt 0" ::: "memory");
    FragBF pf;
#pragma unroll
    for (int v = 0; v < 8; v++)
      pf.u[v] = *(const u32*)(P + l16 * 32 + abase(v) + half * 8);
    // PV: V as B matrix [32 x 16] per d-chunk, contiguous thanks to the transposed store
    size_t vbase = ((size_t)b * HKVn + hkv) * Dh * S;
#pragma unroll
    for (int c = 0; c < 8; c++) {
      FragBF vf;
      int dcol = c * 16 + l16;
#pragma unroll
      for (int v = 0; v < 8; v++)
        vf.u[v] = *(const u32*)(vt + vbase + (size_t)dcol * S + j0 + abase(v) + half * 8);
      ao[c].v = wmma_bf16(pf.v, vf.v, ao[c].v);
    }
  }
#pragma unroll
  for (int c = 0; c < 8; c++)
#pragma unroll
    for (int r = 0; r < 8; r++) {
      int m = i0 + r + half * 8;
      int d = c * 16 + l16;
      o[((size_t)(b * S + m) * HQn + h) * Dh + d] = f2bf(ao[c].f[r] / rowsum[r]);
    }
}

// ===================== LoRA (tiny, VALU) =====================
// t[m][r] = sum_k h2[m][k] * la[k][r]   (K=2048, R=16)
__global__ __launch_bounds__(256) void lora_t_kernel(const u16* __restrict__ h2,
                                                     const float* __restrict__ la,
                                                     float* __restrict__ t) {
  __shared__ float red[256];
  int m = blockIdx.x;
  int tid = threadIdx.x;
  int r = tid & 15, slice = tid >> 4;                  // 16 K-slices of 128
  float s = 0.0f;
  int kbeg = slice * 128;
  for (int kk = kbeg; kk < kbeg + 128; kk++)
    s += bf2f(h2[(size_t)m * 2048 + kk]) * la[(size_t)kk * 16 + r];
  red[tid] = s;
  __syncthreads();
  if (tid < 16) {
    float acc = 0.0f;
#pragma unroll
    for (int i = 0; i < 16; i++) acc += red[i * 16 + tid];
    t[(size_t)m * 16 + tid] = acc;
  }
}

// out[m][n] += (ALPHA/R) * sum_r t[m][r] * lb[r][n]
__global__ __launch_bounds__(256) void lora_final_kernel(const float* __restrict__ t,
                                                         const float* __restrict__ lb,
                                                         float* __restrict__ out) {
  int n = blockIdx.x * 256 + threadIdx.x;
  int m = blockIdx.y;
  float acc = 0.0f;
#pragma unroll
  for (int r = 0; r < 16; r++) acc += t[(size_t)m * 16 + r] * lb[(size_t)r * 2048 + n];
  out[(size_t)m * 2048 + n] += 2.0f * acc;             // ALPHA/R = 32/16
}

// ===================== launch =====================
extern "C" void kernel_launch(void* const* d_in, const int* in_sizes, int n_in,
                              void* d_out, int out_size, void* d_ws, size_t ws_size,
                              hipStream_t stream) {
  const float* x     = (const float*)d_in[0];
  const float* anw   = (const float*)d_in[1];
  const float* wq    = (const float*)d_in[2];
  const float* wk    = (const float*)d_in[3];
  const float* wvv   = (const float*)d_in[4];
  const float* wo    = (const float*)d_in[5];
  const float* mnw   = (const float*)d_in[6];
  const float* wgate = (const float*)d_in[7];
  const float* wup   = (const float*)d_in[8];
  const float* wdown = (const float*)d_in[9];
  const float* la    = (const float*)d_in[10];
  const float* lb    = (const float*)d_in[11];
  float* out = (float*)d_out;
  (void)in_sizes; (void)n_in; (void)out_size; (void)ws_size;

  const int Mtok = 2 * 2048;                           // B*S = 4096
  char* wsp = (char*)d_ws;
  size_t off = 0;
  auto walloc = [&](size_t bytes) -> void* {
    void* p = wsp + off;
    off += (bytes + 255) & ~(size_t)255;
    return p;
  };
  u16* h1   = (u16*)walloc((size_t)Mtok * 2048 * 2);   // normed activations (bf16)
  u16* qb   = (u16*)walloc((size_t)Mtok * 2048 * 2);   // q  [B,S,HQ,D]
  u16* kb   = (u16*)walloc((size_t)Mtok * 512 * 2);    // k  [B,S,HKV,D]
  u16* vtb  = (u16*)walloc((size_t)Mtok * 512 * 2);    // vt [B,HKV,D,S]
  u16* attn = (u16*)walloc((size_t)Mtok * 2048 * 2);   // attention out (bf16)
  u16* h2   = (u16*)walloc((size_t)Mtok * 2048 * 2);   // mlp-normed (bf16)
  u16* act  = (u16*)walloc((size_t)Mtok * 8192 * 2);   // silu(g)*u (bf16)
  float* tl = (float*)walloc((size_t)Mtok * 16 * 4);   // lora intermediate

  dim3 blk(256);
  rmsnorm_kernel<<<Mtok, blk, 0, stream>>>(x, anw, h1);
  gemm_kernel<0><<<dim3(2048 / 128, Mtok / 128), blk, 0, stream>>>(h1, wq, qb, nullptr, Mtok, 2048, 2048, 0);
  gemm_kernel<0><<<dim3(512 / 128, Mtok / 128), blk, 0, stream>>>(h1, wk, kb, nullptr, Mtok, 512, 2048, 0);
  gemm_kernel<1><<<dim3(512 / 128, Mtok / 128), blk, 0, stream>>>(h1, wvv, vtb, nullptr, Mtok, 512, 2048, 2048);
  {
    int tq = Mtok * 16 * 64;
    rope_kernel<<<(tq + 255) / 256, blk, 0, stream>>>(qb, 16, tq);
    int tk = Mtok * 4 * 64;
    rope_kernel<<<(tk + 255) / 256, blk, 0, stream>>>(kb, 4, tk);
  }
  attn_kernel<<<dim3(2048 / 64, 16, 2), dim3(128), 0, stream>>>(qb, kb, vtb, attn);
  gemm_kernel<2><<<dim3(2048 / 128, Mtok / 128), blk, 0, stream>>>(attn, wo, out, x, Mtok, 2048, 2048, 0);
  rmsnorm_kernel<<<Mtok, blk, 0, stream>>>(out, mnw, h2);
  gemm_gateup_kernel<<<dim3(8192 / 128, Mtok / 128), blk, 0, stream>>>(h2, wgate, wup, act, Mtok, 8192, 2048);
  gemm_kernel<3><<<dim3(2048 / 128, Mtok / 128), blk, 0, stream>>>(act, wdown, out, nullptr, Mtok, 2048, 8192, 0);
  lora_t_kernel<<<Mtok, blk, 0, stream>>>(h2, la, tl);
  lora_final_kernel<<<dim3(2048 / 256, Mtok), blk, 0, stream>>>(tl, lb, out);
}
